// DevignModel_807453851673
// MI455X (gfx1250) — matrix operand order repaired and checked by reading, not compiled, not run on hardware
//
#include <hip/hip_runtime.h>
#include <hip/hip_bf16.h>
#include <stdint.h>

// CDNA5 wave32 WMMA types
typedef __attribute__((ext_vector_type(16))) __bf16 v16bf;
typedef __attribute__((ext_vector_type(8)))  float  v8f;

#define WPB 8          // waves per 256-thread block
#define TB  256
#define GSTRIDE 388    // LDS row stride (floats) for 384-wide gate tiles; avoids 64-bank conflicts

__device__ __forceinline__ unsigned short f2bf_bits(float f) {
  unsigned u = __float_as_uint(f);
  unsigned r = u + 0x7FFFu + ((u >> 16) & 1u);   // round-to-nearest-even
  return (unsigned short)(r >> 16);
}

// ---------------- pack A: f32 [N,128] -> bf16 WMMA-A tiles [ntiles][4 kc][32 lanes][16]
// ISA 16-bit A 16x32 layout: lane M=lane&15, group g=lane>>4,
// element jj -> VGPR v=jj>>1 : K = (v<4?0:16) + g*8 + (v&3)*2 + (jj&1)
__global__ void packA_kernel(const float* __restrict__ src, unsigned short* __restrict__ dst,
                             int N, int ntiles) {
  int tid = blockIdx.x * blockDim.x + threadIdx.x;
  int total = ntiles * 4 * 32;
  if (tid >= total) return;
  int lane = tid & 31;
  int kc   = (tid >> 5) & 3;
  int t    = tid >> 7;
  int M = lane & 15, g = lane >> 4;
  int row = t * 16 + M;
  unsigned short* d = dst + (size_t)tid * 16;
  const float* s = src + (size_t)row * 128 + kc * 32;
#pragma unroll
  for (int jj = 0; jj < 16; ++jj) {
    int v = jj >> 1;
    int K = ((v < 4) ? 0 : 16) + g * 8 + (v & 3) * 2 + (jj & 1);
    float f = (row < N) ? s[K] : 0.0f;
    d[jj] = f2bf_bits(f);
  }
}

// ---------------- pack B: f32 weights -> bf16 WMMA-B tiles [nts][kcs][32][16]
// ISA 16-bit B KxN layout: n = lane&15, k-group = lane>>4 within a 32-K chunk,
// element jj -> k = kc*32 + (jj<8?0:16) + (lane>>4)*8 + (jj&7)
__global__ void packB_kernel(const float* __restrict__ src, unsigned short* __restrict__ dst,
                             int K, int Ntot, int transpose) {
  int tid = blockIdx.x * blockDim.x + threadIdx.x;
  int kcs = K >> 5;
  int nts = Ntot >> 4;
  int total = nts * kcs * 32;
  if (tid >= total) return;
  int lane = tid & 31;
  int kc   = (tid >> 5) % kcs;
  int nt   = tid / (32 * kcs);
  int g = lane >> 4;
  int n = nt * 16 + (lane & 15);
  unsigned short* d = dst + (size_t)tid * 16;
#pragma unroll
  for (int jj = 0; jj < 16; ++jj) {
    int k = kc * 32 + ((jj < 8) ? 0 : 16) + g * 8 + (jj & 7);
    float f = transpose ? src[(size_t)n * K + k] : src[(size_t)k * Ntot + n];
    d[jj] = f2bf_bits(f);
  }
}

// ---------------- C = A_tile @ B  (K=128 -> 4 WMMAs/tile); C rows padded to ntiles*16
// LDC compile-time so row stores use immediate offsets in one store clause.
template <int LDC>
__global__ __launch_bounds__(256) void wmma_gemm_kernel(
    const unsigned short* __restrict__ PA, const unsigned short* __restrict__ PB,
    float* __restrict__ C, int ntN) {
  int t    = blockIdx.x;
  int wave = threadIdx.x >> 5;
  int lane = threadIdx.x & 31;
  const v16bf* pa = (const v16bf*)(PA + (size_t)t * 4 * 32 * 16);
  // A tile resident across the whole nt loop
  v16bf a0 = pa[0 * 32 + lane];
  v16bf a1 = pa[1 * 32 + lane];
  v16bf a2 = pa[2 * 32 + lane];
  v16bf a3 = pa[3 * 32 + lane];
  for (int nt = wave; nt < ntN; nt += WPB) {
    const v16bf* pb = (const v16bf*)(PB + (size_t)nt * 4 * 32 * 16);
    // independent loads -> one wide clause, overlap with WMMA issue
    v16bf b0 = pb[0 * 32 + lane];
    v16bf b1 = pb[1 * 32 + lane];
    v16bf b2 = pb[2 * 32 + lane];
    v16bf b3 = pb[3 * 32 + lane];
    v8f acc = {};
    acc = __builtin_amdgcn_wmma_f32_16x16x32_bf16(false, a0, false, b0, (short)0, acc, false, false);
    acc = __builtin_amdgcn_wmma_f32_16x16x32_bf16(false, a1, false, b1, (short)0, acc, false, false);
    acc = __builtin_amdgcn_wmma_f32_16x16x32_bf16(false, a2, false, b2, (short)0, acc, false, false);
    acc = __builtin_amdgcn_wmma_f32_16x16x32_bf16(false, a3, false, b3, (short)0, acc, false, false);
    int col = nt * 16 + (lane & 15);
    int g = lane >> 4;
    float* cp = C + (size_t)(t * 16 + g * 8) * LDC + col;
#pragma unroll
    for (int r = 0; r < 8; ++r)        // C/D layout: VGPR r, lanes16-31 -> M=8+r
      cp[r * LDC] = acc[r];            // constant byte offsets -> store clause
  }
}

// ---------------- edge scatter-add: agg[dst] += m[src], one wave per edge
__global__ void scatter_add_kernel(const float* __restrict__ m, const int* __restrict__ srcI,
                                   const int* __restrict__ dstI, float* __restrict__ agg,
                                   int E) {
  int tid = blockIdx.x * blockDim.x + threadIdx.x;
  int e = tid >> 5;
  if (e >= E) return;
  int lane = tid & 31;
  int s = srcI[e], d = dstI[e];
  const float* mp = m + (size_t)s * 128;
  float* ap = agg + (size_t)d * 128;
#pragma unroll
  for (int p = 0; p < 4; ++p) {
    int c = lane + p * 32;
    atomicAdd(&ap[c], mp[c]);
  }
}

// ---------------- fused GRU: gi = agg@Wih.T+bih, gh = h@Whh.T+bhh (WMMA into LDS), gates, h update
__global__ __launch_bounds__(256) void gru_fused_kernel(
    const unsigned short* __restrict__ PAagg, const unsigned short* __restrict__ PAh,
    const unsigned short* __restrict__ PBih,  const unsigned short* __restrict__ PBhh,
    const float* __restrict__ bih, const float* __restrict__ bhh,
    float* __restrict__ h, int N) {
  __shared__ float gi[16 * GSTRIDE];
  __shared__ float gh[16 * GSTRIDE];
  int t    = blockIdx.x;
  int wave = threadIdx.x >> 5;
  int lane = threadIdx.x & 31;
  // 48 output tiles (24 for gi, 24 for gh) spread over 8 waves
  for (int job = wave; job < 48; job += WPB) {
    int isGI = job < 24;
    int nt = isGI ? job : job - 24;
    const unsigned short* PA = isGI ? PAagg : PAh;
    const unsigned short* PB = isGI ? PBih  : PBhh;
    const float* bias = isGI ? bih : bhh;
    float* out = isGI ? gi : gh;
    const v16bf* pa = (const v16bf*)(PA + (size_t)t * 4 * 32 * 16);
    const v16bf* pb = (const v16bf*)(PB + (size_t)nt * 4 * 32 * 16);
    v16bf a0 = pa[0 * 32 + lane];
    v16bf a1 = pa[1 * 32 + lane];
    v16bf a2 = pa[2 * 32 + lane];
    v16bf a3 = pa[3 * 32 + lane];
    v16bf b0 = pb[0 * 32 + lane];
    v16bf b1 = pb[1 * 32 + lane];
    v16bf b2 = pb[2 * 32 + lane];
    v16bf b3 = pb[3 * 32 + lane];
    v8f acc = {};
    acc = __builtin_amdgcn_wmma_f32_16x16x32_bf16(false, a0, false, b0, (short)0, acc, false, false);
    acc = __builtin_amdgcn_wmma_f32_16x16x32_bf16(false, a1, false, b1, (short)0, acc, false, false);
    acc = __builtin_amdgcn_wmma_f32_16x16x32_bf16(false, a2, false, b2, (short)0, acc, false, false);
    acc = __builtin_amdgcn_wmma_f32_16x16x32_bf16(false, a3, false, b3, (short)0, acc, false, false);
    int col = nt * 16 + (lane & 15);
    int g = lane >> 4;
    float bv = bias[col];
#pragma unroll
    for (int r = 0; r < 8; ++r)
      out[(g * 8 + r) * GSTRIDE + col] = acc[r] + bv;
  }
  __syncthreads();
  for (int i = threadIdx.x; i < 16 * 128; i += TB) {
    int row = i >> 7, c = i & 127;
    int node = t * 16 + row;
    if (node >= N) continue;
    float ir = gi[row * GSTRIDE + c], iz = gi[row * GSTRIDE + 128 + c], in_ = gi[row * GSTRIDE + 256 + c];
    float hr = gh[row * GSTRIDE + c], hz = gh[row * GSTRIDE + 128 + c], hn  = gh[row * GSTRIDE + 256 + c];
    float rg = 1.0f / (1.0f + __expf(-(ir + hr)));
    float zg = 1.0f / (1.0f + __expf(-(iz + hz)));
    float ng = tanhf(in_ + rg * hn);
    float hold = h[(size_t)node * 128 + c];
    h[(size_t)node * 128 + c] = (1.0f - zg) * ng + zg * hold;
  }
}

// ---------------- segment-max via monotonic uint encoding + atomicMax
__global__ void fill_u32_kernel(unsigned* __restrict__ p, unsigned v, int n) {
  int tid = blockIdx.x * blockDim.x + threadIdx.x;
  if (tid < n) p[tid] = v;
}

__global__ void segmax_kernel(const float* __restrict__ h, const int* __restrict__ batch,
                              unsigned* __restrict__ keys, int N) {
  int tid = blockIdx.x * blockDim.x + threadIdx.x;
  if (tid >= N * 128) return;
  int node = tid >> 7, c = tid & 127;
  float f = h[(size_t)node * 128 + c];
  unsigned u = __float_as_uint(f);
  unsigned key = ((int)u < 0) ? ~u : (u | 0x80000000u);
  atomicMax(&keys[(size_t)batch[node] * 128 + c], key);
}

__global__ void segmax_decode_kernel(const unsigned* __restrict__ keys,
                                     float* __restrict__ out, int n) {
  int tid = blockIdx.x * blockDim.x + threadIdx.x;
  if (tid >= n) return;
  unsigned key = keys[tid];
  unsigned u = (key & 0x80000000u) ? (key & 0x7FFFFFFFu) : ~key;
  out[tid] = __uint_as_float(u);
}

// ---------------- small dense layer: out[g][o] = act(in[g]·W[o] + b[o])
__global__ void dense_kernel(const float* __restrict__ in, const float* __restrict__ W,
                             const float* __restrict__ b, float* __restrict__ out,
                             int Gn, int I, int O, int relu) {
  int tid = blockIdx.x * blockDim.x + threadIdx.x;
  if (tid >= Gn * O) return;
  int g = tid / O, o = tid % O;
  const float* ip = in + (size_t)g * I;
  const float* wp = W + (size_t)o * I;
  float s = b[o];
  for (int i = 0; i < I; ++i) s += ip[i] * wp[i];
  if (relu) s = fmaxf(s, 0.0f);
  out[(size_t)g * O + o] = s;
}

static inline size_t alignUp(size_t x, size_t a) { return (x + a - 1) & ~(a - 1); }

extern "C" void kernel_launch(void* const* d_in, const int* in_sizes, int n_in,
                              void* d_out, int out_size, void* d_ws, size_t ws_size,
                              hipStream_t stream) {
  const float* x        = (const float*)d_in[0];
  const int*   eidx     = (const int*)d_in[1];
  const int*   batch    = (const int*)d_in[2];
  const float* ggnn_w   = (const float*)d_in[3];   // [5,128,128]
  const float* gru_wih  = (const float*)d_in[4];   // [384,128]
  const float* gru_whh  = (const float*)d_in[5];   // [384,128]
  const float* gru_bih  = (const float*)d_in[6];   // [384]
  const float* gru_bhh  = (const float*)d_in[7];   // [384]
  const float* W1  = (const float*)d_in[8];  const float* b1  = (const float*)d_in[9];
  const float* Wf1 = (const float*)d_in[10]; const float* bf1 = (const float*)d_in[11];
  const float* Wf2 = (const float*)d_in[12]; const float* bf2 = (const float*)d_in[13];
  const float* Wc  = (const float*)d_in[14]; const float* bc  = (const float*)d_in[15];
  float* out = (float*)d_out;

  const int D  = 128;
  const int N  = in_sizes[0] / D;
  const int E  = in_sizes[1] / 2;
  const int Gn = out_size / 2;          // num graphs
  const int H1 = in_sizes[9];           // 256
  const int H2 = in_sizes[11];          // 128
  const int L  = in_sizes[3] / (D * D); // 5
  const int ntiles = (N + 15) / 16;
  const size_t Npad = (size_t)ntiles * 16;

  // ---- carve workspace
  char* p = (char*)d_ws;
  size_t off = 0;
  auto carve = [&](size_t bytes) -> char* {
    char* r = p + off;
    off = alignUp(off + bytes, 256);
    return r;
  };
  float* h   = (float*)carve(Npad * D * 4);
  float* m   = (float*)carve(Npad * D * 4);
  float* agg = (float*)carve(Npad * D * 4);
  unsigned short* PAh = (unsigned short*)carve((size_t)ntiles * 4 * 32 * 16 * 2);
  unsigned short* PAa = (unsigned short*)carve((size_t)ntiles * 4 * 32 * 16 * 2);
  unsigned short* PBg  = (unsigned short*)carve((size_t)L * 8 * 4 * 32 * 16 * 2);   // 5x 128x128
  unsigned short* PBih = (unsigned short*)carve((size_t)24 * 4 * 32 * 16 * 2);      // K=128,N=384
  unsigned short* PBhh = (unsigned short*)carve((size_t)24 * 4 * 32 * 16 * 2);
  unsigned* keys = (unsigned*)carve((size_t)Gn * D * 4);
  float* gemb = (float*)carve((size_t)Gn * D * 4);
  float* mb1  = (float*)carve((size_t)Gn * H1 * 4);
  float* mb2  = (float*)carve((size_t)Gn * H2 * 4);
  float* mb3  = (float*)carve((size_t)Gn * H1 * 4);
  (void)ws_size; (void)n_in;

  // ---- h = x
  hipMemcpyAsync(h, x, (size_t)N * D * sizeof(float), hipMemcpyDeviceToDevice, stream);

  // ---- pack all weights to bf16 WMMA-B layout (once per call; tiny)
  {
    int totB128 = 8 * 4 * 32;   // per 128x128 matrix
    for (int l = 0; l < L; ++l)
      packB_kernel<<<(totB128 + TB - 1) / TB, TB, 0, stream>>>(
          ggnn_w + (size_t)l * D * D, PBg + (size_t)l * 8 * 4 * 32 * 16, D, D, 0);
    int totB384 = 24 * 4 * 32;
    packB_kernel<<<(totB384 + TB - 1) / TB, TB, 0, stream>>>(gru_wih, PBih, D, 3 * D, 1);
    packB_kernel<<<(totB384 + TB - 1) / TB, TB, 0, stream>>>(gru_whh, PBhh, D, 3 * D, 1);
  }

  const int packTot = ntiles * 4 * 32;
  const int packBlocks = (packTot + TB - 1) / TB;
  const int scatTot = E * 32;
  const int scatBlocks = (scatTot + TB - 1) / TB;

  // ---- GGNN layers
  for (int l = 0; l < L; ++l) {
    packA_kernel<<<packBlocks, TB, 0, stream>>>(h, PAh, N, ntiles);
    wmma_gemm_kernel<128><<<ntiles, TB, 0, stream>>>(
        PAh, PBg + (size_t)l * 8 * 4 * 32 * 16, m, 8);
    hipMemsetAsync(agg, 0, Npad * D * sizeof(float), stream);
    scatter_add_kernel<<<scatBlocks, TB, 0, stream>>>(m, eidx, eidx + E, agg, E);
    packA_kernel<<<packBlocks, TB, 0, stream>>>(agg, PAa, N, ntiles);
    gru_fused_kernel<<<ntiles, TB, 0, stream>>>(PAa, PAh, PBih, PBhh,
                                                gru_bih, gru_bhh, h, N);
  }

  // ---- graph readout: segment max
  {
    int nk = Gn * D;
    fill_u32_kernel<<<(nk + TB - 1) / TB, TB, 0, stream>>>(keys, 0x007FFFFFu, nk); // enc(-inf)
    int nt2 = N * D;
    segmax_kernel<<<(nt2 + TB - 1) / TB, TB, 0, stream>>>(h, batch, keys, N);
    segmax_decode_kernel<<<(nk + TB - 1) / TB, TB, 0, stream>>>(keys, gemb, nk);
  }

  // ---- MLP head (tiny)
  dense_kernel<<<(Gn * H1 + TB - 1) / TB, TB, 0, stream>>>(gemb, W1,  b1,  mb1, Gn, D,  H1, 1);
  dense_kernel<<<(Gn * H2 + TB - 1) / TB, TB, 0, stream>>>(mb1,  Wf1, bf1, mb2, Gn, H1, H2, 1);
  dense_kernel<<<(Gn * H1 + TB - 1) / TB, TB, 0, stream>>>(mb2,  Wf2, bf2, mb3, Gn, H2, H1, 1);
  dense_kernel<<<(Gn * 2  + TB - 1) / TB, TB, 0, stream>>>(mb3,  Wc,  bc,  out, Gn, H1, 2,  0);
}
